// LearnableGlobalLocalMultiheadAttention_1451698946736
// MI455X (gfx1250) — matrix operand order repaired and checked by reading, not compiled, not run on hardware
//
#include <hip/hip_runtime.h>
#include <cstdint>

// ---------------------------------------------------------------------------
// LearnableGlobalLocalMultiheadAttention for MI455X (gfx1250, wave32, WMMA)
//
// The four [32,1024,1024]x[1024,1024] mask GEMMs are replaced by 2D
// prefix/suffix scans (triu mask is separable: (k<=j) & (k%p <= j%p)),
// cutting ~275 of ~290 GFLOP. All remaining GEMMs use
// v_wmma_f32_16x16x32_f16 (f16 in, f32 accumulate).  The V operand of the
// attn_w @ v GEMM is staged into LDS with the Tensor Data Mover
// (tensor_load_to_lds, TENSORcnt), overlapping the score/softmax phases.
// ---------------------------------------------------------------------------

typedef _Float16 h16;
typedef __attribute__((ext_vector_type(16))) _Float16 v16h;
typedef __attribute__((ext_vector_type(8)))  _Float16 v8h;
typedef __attribute__((ext_vector_type(8)))  float    v8f;

typedef unsigned int u32x4 __attribute__((ext_vector_type(4)));
typedef int          i32x8 __attribute__((ext_vector_type(8)));
typedef int          i32x4 __attribute__((ext_vector_type(4)));

typedef __attribute__((address_space(3))) _Float16 lds_h16;

#define SEQ   1024
#define BSZ   4
#define EMB   256
#define NHEAD 8
#define HDIM  32
#define NBH   32           // BSZ * NHEAD
#define SCALING 0.17677669529663687f   // 32^-0.5

#if defined(__has_builtin)
# if __has_builtin(__builtin_amdgcn_tensor_load_to_lds)
#  define HAVE_TDM 1
# endif
#endif
#ifndef HAVE_TDM
# define HAVE_TDM 0
#endif

// ---------------- WMMA fragment helpers (CDNA5 wave32 layouts) -------------

__device__ __forceinline__ int lane_id() { return (int)(threadIdx.x & 31u); }

// A fragment (16 rows x 32 K) from a row-major f16 matrix, leading dim ld.
// Lane L holds row M=L&15; half j -> K = (L>>4)*16 + j.
// Two contiguous 16B loads per lane.
__device__ __forceinline__ v16h frag_a_rowmajor(const h16* base, int ld) {
  const int l = lane_id();
  const int m = l & 15, hf = l >> 4;
  const h16* r = base + m * ld + hf * 16;
  v8h lo = *(const v8h*)(r);
  v8h hi = *(const v8h*)(r + 8);
  v16h a;
#pragma unroll
  for (int j = 0; j < 8; ++j) { a[j] = lo[j]; a[8 + j] = hi[j]; }
  return a;
}

// B fragment (32 K x 16 cols) where B[k][n] = M[n][k] for row-major M
// (i.e. B = M^T).  Lane L holds col n=L&15; half j -> K = (L>>4)*16 + j.
// One contiguous 32B load per lane.
__device__ __forceinline__ v16h frag_b_transposed(const h16* base, int ld) {
  const int l = lane_id();
  const int n = l & 15, hf = l >> 4;
  return *(const v16h*)(base + n * ld + hf * 16);
}

// B fragment from a row-major M[k][n] (no transpose): strided gathers
// (used only on LDS-resident data -> ds_load).
__device__ __forceinline__ v16h frag_b_rowmajor(const h16* base, int ld) {
  const int l = lane_id();
  const int n = l & 15, hf = l >> 4;
  v16h b;
#pragma unroll
  for (int j = 0; j < 16; ++j) b[j] = base[(hf * 16 + j) * ld + n];
  return b;
}

__device__ __forceinline__ v8f wmma_f16(v16h a, v16h b, v8f c) {
  return __builtin_amdgcn_wmma_f32_16x16x32_f16(false, a, false, b,
                                                (short)0, c, false, false);
}

// ---------------- TDM helpers ----------------------------------------------

#if HAVE_TDM
// Issue one TDM descriptor copying `nelem` f16 (contiguous) from global to
// LDS.  D# per CDNA5 ISA: 1-row 2D tile,
// data_size=2B, tile_dim0=nelem.  Wave-level op; EXEC ignored; TENSORcnt.
__device__ __forceinline__ void tdm_load_f16(const h16* g, uint32_t lds_off,
                                             uint32_t nelem) {
  const uint64_t ga = (uint64_t)(uintptr_t)g;
  u32x4 g0;
  g0[0] = 1u;                                    // count=1 (valid user D#)
  g0[1] = lds_off;                               // lds_addr (bytes)
  g0[2] = (uint32_t)ga;                          // global_addr[31:0]
  g0[3] = (uint32_t)((ga >> 32) & 0x01FFFFFFull) // global_addr[56:32]
        | (2u << 30);                            // type=2 ("image")
  i32x8 g1;
  g1[0] = 0x00010000;                            // data_size=1 (2 bytes)
  g1[1] = (int)((nelem & 0xFFFFu) << 16);        // tensor_dim0[15:0]
  g1[2] = (int)((nelem >> 16) | (1u << 16));     // tensor_dim0[31:16] | dim1=1
  g1[3] = (int)((nelem & 0xFFFFu) << 16);        // tile_dim0 = nelem
  g1[4] = 0;                                     // tile_dim1/2 unused
  g1[5] = (int)nelem;                            // tensor_dim0_stride[31:0]
  g1[6] = 0;
  g1[7] = 0;
  i32x4 z4 = {0, 0, 0, 0};
#if __clang_major__ >= 23
  i32x8 z8 = {0, 0, 0, 0, 0, 0, 0, 0};
  __builtin_amdgcn_tensor_load_to_lds(g0, g1, z4, z4, z8, 0);
#else
  __builtin_amdgcn_tensor_load_to_lds(g0, g1, z4, z4, 0);
#endif
}
#endif

__device__ __forceinline__ void wait_tensorcnt0() {
#if defined(__has_builtin)
# if __has_builtin(__builtin_amdgcn_s_wait_tensorcnt)
  __builtin_amdgcn_s_wait_tensorcnt(0);
  return;
# endif
#endif
  asm volatile("s_wait_tensorcnt 0x0" ::: "memory");
}

// ---------------- K0: f32 -> f16 conversion --------------------------------

__global__ void k_cvt(const float* __restrict__ s, h16* __restrict__ d, int n) {
  int i = blockIdx.x * blockDim.x + threadIdx.x;
  if (i < n) d[i] = (h16)s[i];
}

// ---------------- K1: fused 9-way input projection GEMM --------------------
// M = S*B = 4096, N = 9*256 = 2304, K = 256.  One 16x16 tile per wave.
// Output in head-major layout proj[i][bh][s][d], scaling folded into i=0,7.

__global__ void k_proj(const h16* __restrict__ qkv, const h16* __restrict__ wh,
                       const float* __restrict__ bias, h16* __restrict__ proj) {
  const int wave = blockIdx.x * (blockDim.x >> 5) + (threadIdx.x >> 5);
  const int tm = wave & 255;        // 256 row tiles
  const int tn = wave >> 8;         // 144 col tiles
  if (tn >= 144) return;
  const int row0 = tm * 16;
  const int col0 = tn * 16;
  const int pi = col0 >> 8;         // projection slice 0..8
  // 0,3,5,7 -> query; 1,4,6,8 -> key; 2 -> value
  const int sel = (pi == 2) ? 2 : ((pi == 1 || pi == 4 || pi == 6 || pi == 8) ? 1 : 0);
  const h16* X = qkv + (size_t)sel * (SEQ * BSZ * EMB);

  v8f acc = {};
#pragma unroll
  for (int kk = 0; kk < EMB / 32; ++kk) {
    v16h a = frag_a_rowmajor(X + row0 * EMB + kk * 32, EMB);
    v16h b = frag_b_transposed(wh + col0 * EMB + kk * 32, EMB);
    acc = wmma_f16(a, b, acc);
  }
  const float scale = (pi == 0 || pi == 7) ? SCALING : 1.0f;
  const int l = lane_id();
  const int n = l & 15, hf = l >> 4;
  const int col = col0 + n;
  const int c256 = col & 255;
  const int h = c256 >> 5, d = c256 & 31;
#pragma unroll
  for (int r = 0; r < 8; ++r) {
    const int m = row0 + r + 8 * hf;          // row = s*BSZ + b
    const int s = m >> 2, bidx = m & 3;
    float val = (acc[r] + bias[col]) * scale;
    proj[(((size_t)pi * NBH + bidx * NHEAD + h) * SEQ + s) * HDIM + d] = (h16)val;
  }
}

// ---------------- K2+K3 fused: left/right softmax + 2D scans -> local_mask -
// One block per (bh, 16-row slab).  8 waves.
//   phase 1: WMMA scores (hd=32 = exactly one 16x16x32 WMMA per tile)
//   phase 2: row softmax in LDS
//   phase 3: wave-scan 2D prefix/suffix -> mask = P_L*S_R + S_L*P_R

__global__ void k_mask(const h16* __restrict__ proj, h16* __restrict__ mask,
                       const int* __restrict__ shape1p) {
  __shared__ __align__(16) h16 smL[16][SEQ];
  __shared__ __align__(16) h16 smR[16][SEQ];
  const int bh = blockIdx.y;
  const int i0 = blockIdx.x * 16;
  const int wid = threadIdx.x >> 5;
  const int l = lane_id();
  const int n = l & 15, hf = l >> 4;

  const h16* qL = proj + ((size_t)(3 * NBH + bh) * SEQ) * HDIM;
  const h16* kL = proj + ((size_t)(4 * NBH + bh) * SEQ) * HDIM;
  const h16* qR = proj + ((size_t)(5 * NBH + bh) * SEQ) * HDIM;
  const h16* kR = proj + ((size_t)(6 * NBH + bh) * SEQ) * HDIM;

  // phase 1: raw scores, 64 col tiles over 8 waves
  v16h aL = frag_a_rowmajor(qL + i0 * HDIM, HDIM);
  v16h aR = frag_a_rowmajor(qR + i0 * HDIM, HDIM);
  for (int tt = 0; tt < 8; ++tt) {
    const int t = wid * 8 + tt;
    v8f z = {};
    v8f cL = wmma_f16(aL, frag_b_transposed(kL + t * 16 * HDIM, HDIM), z);
    v8f cR = wmma_f16(aR, frag_b_transposed(kR + t * 16 * HDIM, HDIM), z);
#pragma unroll
    for (int r = 0; r < 8; ++r) {
      smL[r + 8 * hf][t * 16 + n] = (h16)cL[r];
      smR[r + 8 * hf][t * 16 + n] = (h16)cR[r];
    }
  }
  __syncthreads();

  // phase 2: softmax; 32 row-jobs (16 left + 16 right), 4 per wave
  for (int job = wid; job < 32; job += 8) {
    h16* row = (job < 16) ? smL[job] : smR[job - 16];
    float mx = -1e30f;
    for (int j = l; j < SEQ; j += 32) mx = fmaxf(mx, (float)row[j]);
#pragma unroll
    for (int o = 16; o; o >>= 1) mx = fmaxf(mx, __shfl_xor(mx, o));
    float sum = 0.f;
    for (int j = l; j < SEQ; j += 32) sum += __expf((float)row[j] - mx);
#pragma unroll
    for (int o = 16; o; o >>= 1) sum += __shfl_xor(sum, o);
    float inv = 1.0f / sum;
    for (int j = l; j < SEQ; j += 32) row[j] = (h16)(__expf((float)row[j] - mx) * inv);
  }
  __syncthreads();

  // phase 3: separable triu mask ->
  //   (L@triu)[ap+b]   = sum_{a'<=a} sum_{c<=b} L[a'p+c]  (2D prefix)
  //   (L@triu_t)[ap+b] = sum_{a'>=a} sum_{c>=b} L[a'p+c]  (2D suffix)
  // mask = P_L*S_R + S_L*P_R.  (p <= 32 maps onto one wave; setup uses p=32.)
  int p = *shape1p; if (p < 1) p = 1; if (p > 32) p = 32;
  const int nblk0 = SEQ / p;
  const int nblk = (nblk0 < 64) ? nblk0 : 64;   // assumes p >= 16 (setup: 32)
  for (int rr = 0; rr < 2; ++rr) {
    const int row = wid * 2 + rr;
    float pl[64], pr[64];
    float accPL = 0.f, accPR = 0.f;
    for (int a = 0; a < nblk; ++a) {
      float wl = (l < p) ? (float)smL[row][a * p + l] : 0.f;
      float wr = (l < p) ? (float)smR[row][a * p + l] : 0.f;
#pragma unroll
      for (int o = 1; o < 32; o <<= 1) {        // inclusive wave scan
        float tl = __shfl_up(wl, o), tr = __shfl_up(wr, o);
        if (l >= o) { wl += tl; wr += tr; }
      }
      accPL += wl; accPR += wr;                 // cross-block prefix
      pl[a] = accPL; pr[a] = accPR;
    }
    float accSL = 0.f, accSR = 0.f;
    h16* out = mask + ((size_t)bh * SEQ + (i0 + row)) * SEQ;
    for (int a = nblk - 1; a >= 0; --a) {
      float xl = (l < p) ? (float)smL[row][a * p + l] : 0.f;
      float xr = (l < p) ? (float)smR[row][a * p + l] : 0.f;
      float il = xl, ir = xr;
#pragma unroll
      for (int o = 1; o < 32; o <<= 1) {
        float tl = __shfl_up(il, o), tr = __shfl_up(ir, o);
        if (l >= o) { il += tl; ir += tr; }
      }
      float totL = __shfl(il, 31), totR = __shfl(ir, 31);
      accSL += totL - il + xl;                  // within-block suffix, accum
      accSR += totR - ir + xr;
      float mval = pl[a] * accSR + accSL * pr[a];
      if (l < p) out[a * p + l] = (h16)mval;
    }
  }
}

// ---------------- K4: fused attention --------------------------------------
// logits = 0.1*(q@k^T) + (q_loc@k_loc^T)*mask ; softmax ; @ v.
// V ([1024,32] f16 = 64KB) is staged into LDS by the TDM at kernel entry,
// overlapping the score-WMMA and softmax phases; waited with s_wait_tensorcnt
// before the attn_w @ v WMMA phase (B fragments then come from LDS).

__global__ void k_attn(const h16* __restrict__ proj, const h16* __restrict__ mask,
                       h16* __restrict__ att) {
  __shared__ __align__(16)  h16 sw[16][SEQ];        // 32KB: logits -> weights
  __shared__ __align__(128) h16 vlds[SEQ * HDIM];   // 64KB: V tile
  const int bh = blockIdx.y;
  const int i0 = blockIdx.x * 16;
  const int wid = threadIdx.x >> 5;
  const int l = lane_id();
  const int n = l & 15, hf = l >> 4;

  const h16* q  = proj + ((size_t)(0 * NBH + bh) * SEQ) * HDIM;
  const h16* k  = proj + ((size_t)(1 * NBH + bh) * SEQ) * HDIM;
  const h16* v  = proj + ((size_t)(2 * NBH + bh) * SEQ) * HDIM;
  const h16* ql = proj + ((size_t)(7 * NBH + bh) * SEQ) * HDIM;
  const h16* kl = proj + ((size_t)(8 * NBH + bh) * SEQ) * HDIM;
  const h16* mrow = mask + ((size_t)bh * SEQ + i0) * SEQ;

#if HAVE_TDM
  if (wid == 0) {                 // one TDM descriptor per workgroup (wave 0)
    uint32_t voff = (uint32_t)(uintptr_t)(lds_h16*)&vlds[0];
    tdm_load_f16(v, voff, SEQ * HDIM);
  }
#else
  for (int idx = threadIdx.x; idx < SEQ * HDIM / 8; idx += blockDim.x)
    ((v8h*)vlds)[idx] = ((const v8h*)v)[idx];
#endif

  v16h aq  = frag_a_rowmajor(q  + i0 * HDIM, HDIM);
  v16h aql = frag_a_rowmajor(ql + i0 * HDIM, HDIM);
  for (int tt = 0; tt < 8; ++tt) {
    const int t = wid * 8 + tt;
    __builtin_prefetch(mrow + (size_t)(8 * hf) * SEQ + t * 16, 0, 0);
    v8f z = {};
    v8f cg = wmma_f16(aq,  frag_b_transposed(k  + t * 16 * HDIM, HDIM), z);
    v8f cl = wmma_f16(aql, frag_b_transposed(kl + t * 16 * HDIM, HDIM), z);
#pragma unroll
    for (int r = 0; r < 8; ++r) {
      float mv = (float)mrow[(size_t)(r + 8 * hf) * SEQ + t * 16 + n];
      sw[r + 8 * hf][t * 16 + n] = (h16)(0.1f * cg[r] + cl[r] * mv);
    }
  }
  __syncthreads();

  for (int job = wid; job < 16; job += 8) {     // softmax, 2 rows/wave
    h16* row = sw[job];
    float mx = -1e30f;
    for (int j = l; j < SEQ; j += 32) mx = fmaxf(mx, (float)row[j]);
#pragma unroll
    for (int o = 16; o; o >>= 1) mx = fmaxf(mx, __shfl_xor(mx, o));
    float sum = 0.f;
    for (int j = l; j < SEQ; j += 32) sum += __expf((float)row[j] - mx);
#pragma unroll
    for (int o = 16; o; o >>= 1) sum += __shfl_xor(sum, o);
    float inv = 1.0f / sum;
    for (int j = l; j < SEQ; j += 32) row[j] = (h16)(__expf((float)row[j] - mx) * inv);
  }
#if HAVE_TDM
  if (wid == 0) wait_tensorcnt0();  // V tile landed in LDS
#endif
  __syncthreads();

  if (wid < 2) {                                 // attn_w @ v : N=32 -> 2 tiles
    const int d0 = wid * 16;
    v8f acc = {};
    for (int kk = 0; kk < SEQ / 32; ++kk) {
      v16h a = frag_a_rowmajor(&sw[0][kk * 32], SEQ);      // A from LDS
      v16h b = frag_b_rowmajor(&vlds[kk * 32 * HDIM + d0], HDIM);  // B from LDS
      acc = wmma_f16(a, b, acc);
    }
    const int bidx = bh >> 3, h = bh & 7;
#pragma unroll
    for (int r = 0; r < 8; ++r) {
      const int s = i0 + r + 8 * hf;
      att[((size_t)s * BSZ + bidx) * EMB + h * HDIM + d0 + n] = (h16)acc[r];
    }
  }
}

// ---------------- K5: consistent_mask = sum over heads ---------------------

__global__ void k_consistent(const h16* __restrict__ mask, float* __restrict__ outm) {
  int idx = blockIdx.x * blockDim.x + threadIdx.x;
  if (idx < SEQ * SEQ) {
    float s = 0.f;
#pragma unroll
    for (int bh = 0; bh < NBH; ++bh)
      s += (float)mask[(size_t)bh * SEQ * SEQ + idx];
    outm[idx] = s;
  }
}

// ---------------- K6: output projection ------------------------------------

__global__ void k_outproj(const h16* __restrict__ att, const h16* __restrict__ owh,
                          const float* __restrict__ ob, float* __restrict__ out) {
  const int wave = blockIdx.x * (blockDim.x >> 5) + (threadIdx.x >> 5);
  const int tm = wave & 255, tn = wave >> 8;    // 256 x 16 tiles
  if (tn >= 16) return;
  const int row0 = tm * 16, col0 = tn * 16;
  v8f acc = {};
#pragma unroll
  for (int kk = 0; kk < EMB / 32; ++kk) {
    v16h a = frag_a_rowmajor(att + row0 * EMB + kk * 32, EMB);
    v16h b = frag_b_transposed(owh + col0 * EMB + kk * 32, EMB);
    acc = wmma_f16(a, b, acc);
  }
  const int l = lane_id(), n = l & 15, hf = l >> 4;
  const int col = col0 + n;
#pragma unroll
  for (int r = 0; r < 8; ++r) {
    const int m = row0 + r + 8 * hf;
    out[(size_t)m * EMB + col] = acc[r] + ob[col];
  }
}

// ---------------- launch ----------------------------------------------------

extern "C" void kernel_launch(void* const* d_in, const int* in_sizes, int n_in,
                              void* d_out, int out_size, void* d_ws, size_t ws_size,
                              hipStream_t stream) {
  (void)in_sizes; (void)n_in; (void)out_size; (void)ws_size;
  const float* q  = (const float*)d_in[0];
  const float* kk = (const float*)d_in[1];
  const float* vv = (const float*)d_in[2];
  const float* w  = (const float*)d_in[3];
  const float* b  = (const float*)d_in[4];
  const float* ow = (const float*)d_in[5];
  const float* ob = (const float*)d_in[6];
  const int* sh1  = (const int*)d_in[8];

  // workspace layout (f16 elements); total ~91.3 MB
  h16* ws    = (h16*)d_ws;
  h16* qkv   = ws;                       // 3 * 1048576
  h16* wh    = qkv + 3u * 1048576u;      // 589824
  h16* owh   = wh + 589824u;             // 65536
  h16* proj  = owh + 65536u;             // 9 * 32 * 1024 * 32 = 9437184
  h16* maskw = proj + 9437184u;          // 32 * 1024 * 1024 = 33554432
  h16* att   = maskw + 33554432u;        // 1048576

  const int T = 256;
  k_cvt<<<(1048576 + T - 1) / T, T, 0, stream>>>(q,  qkv,                 1048576);
  k_cvt<<<(1048576 + T - 1) / T, T, 0, stream>>>(kk, qkv + 1048576u,      1048576);
  k_cvt<<<(1048576 + T - 1) / T, T, 0, stream>>>(vv, qkv + 2u * 1048576u, 1048576);
  k_cvt<<<(589824 + T - 1) / T, T, 0, stream>>>(w,  wh,  589824);
  k_cvt<<<(65536  + T - 1) / T, T, 0, stream>>>(ow, owh, 65536);

  k_proj<<<4608, 256, 0, stream>>>(qkv, wh, b, proj);               // 36864 waves
  k_mask<<<dim3(64, 32), 256, 0, stream>>>(proj, maskw, sh1);
  k_attn<<<dim3(64, 32), 256, 0, stream>>>(proj, maskw, att);
  k_consistent<<<4096, 256, 0, stream>>>(maskw, (float*)d_out + 1048576);
  k_outproj<<<512, 256, 0, stream>>>(att, owh, ob, (float*)d_out);
}
